// Attention_5789615915550
// MI455X (gfx1250) — matrix operand order repaired and checked by reading, not compiled, not run on hardware
//
#include <hip/hip_runtime.h>
#include <hip/hip_bf16.h>

// B=1024, L=256, D=256, H=512, U=512
#define B_ 1024
#define L_ 256
#define D_ 256
#define H_ 512
#define U_ 512

typedef __attribute__((ext_vector_type(16))) __bf16 v16bf;
typedef __attribute__((ext_vector_type(8)))  float  v8f;

// Exact parameter types of the gfx1250 async-copy builtin, per hipcc's
// diagnostic: GNU vector "int4" pointers in AS1 (global) / AS3 (LDS).
typedef int gnu_v4i __attribute__((vector_size(16)));
typedef __attribute__((address_space(1))) gnu_v4i* as1_v4i_ptr;
typedef __attribute__((address_space(3))) gnu_v4i* as3_v4i_ptr;

// CDNA5 async global->LDS copy (ASYNCcnt-tracked, no VGPR transit).
#if defined(__has_builtin)
#if __has_builtin(__builtin_amdgcn_global_load_async_to_lds_b128) && \
    __has_builtin(__builtin_amdgcn_s_wait_asynccnt)
#define USE_ASYNC_LDS 1
#endif
#endif

__device__ __forceinline__ float fast_tanh(float x) {
#if defined(__has_builtin) && __has_builtin(__builtin_amdgcn_tanhf)
    return __builtin_amdgcn_tanhf(x);          // native v_tanh_f32 (TRANS32)
#else
    float e = __expf(2.0f * x);
    return 1.0f - 2.0f / (e + 1.0f);
#endif
}

// ---------------------------------------------------------------------------
// Pack W1 (f32, [D,U] row-major) into bf16 in this code's WMMA B-fragment
// layout for V_WMMA_F32_16X16X32_BF16 (contiguous k = 16*half + element, same order as A):
//   lane 0-15  : N = lane,    VGPR r holds K = 2r, 2r+1
//   lane 16-31 : N = lane-16, VGPR r holds K = 16+2r, 17+2r
// Storage: W1p[((nt*8 + kt)*32 + lane)*16 + 2r + s]
// ---------------------------------------------------------------------------
__global__ void pack_w1_kernel(const float* __restrict__ W1,
                               __bf16* __restrict__ W1p) {
    int t = blockIdx.x * blockDim.x + threadIdx.x;   // 0 .. 131071
    int idx  = t & 15;
    int lane = (t >> 4) & 31;
    int kt   = (t >> 9) & 7;
    int nt   = t >> 12;
    int r = idx >> 1, s = idx & 1;
    int k = kt * 32 + ((lane >> 4) << 4) + (r << 1) + s;
    int n = (nt << 4) + (lane & 15);
    W1p[t] = (__bf16)W1[k * U_ + n];
}

// ---------------------------------------------------------------------------
// One workgroup per batch element b. 512 threads = 16 waves (wave32).
// Wave w owns L-rows [16w, 16w+16). A fragments live in registers; B slabs
// are double-buffered through LDS via async copies shared by all 16 waves.
// Logit partials accumulate in registers; one butterfly reduction at the end.
// ---------------------------------------------------------------------------
__launch_bounds__(512)
__global__ void attn_fused_kernel(const float* __restrict__ feat,   // [B,L,D]
                                  const float* __restrict__ hidden, // [B,H]
                                  const float* __restrict__ b1,     // [U]
                                  const float* __restrict__ W2,     // [H,U]
                                  const float* __restrict__ b2,     // [U]
                                  const float* __restrict__ Wv,     // [U,1]
                                  const float* __restrict__ bv,     // [1]
                                  const __bf16* __restrict__ W1p,   // packed
                                  float* __restrict__ out) {
    const int b    = blockIdx.x;
    const int tid  = threadIdx.x;
    const int lane = tid & 31;
    const int wave = tid >> 5;        // 0..15
    const int hi   = lane >> 4;       // half-wave select
    const int m15  = lane & 15;

    __shared__ __align__(32) __bf16 sB[2][8 * 512]; // double-buffered 8KB slabs
    __shared__ float sH[H_];
    __shared__ float sPH[U_];        // b1[u] + b2[u] + hidden[b]@W2
    __shared__ float sWv[U_];
    __shared__ float sLogit[L_];
    __shared__ float sW[L_];
    __shared__ float sRed[16];
    __shared__ float sMax, sSum;
    __shared__ float sCtx[2][D_];

    const float* fb = feat + (size_t)b * (L_ * D_);

    // ---- stage per-b vectors ----
    sH[tid]  = hidden[b * H_ + tid];          // H_ == blockDim
    sWv[tid] = Wv[tid];                       // U_ == blockDim
    __syncthreads();

#ifdef USE_ASYNC_LDS
    // kick off slab 0 copy; it overlaps proj_h + A-fragment conversion below
    {
        const __bf16* gsrc = W1p + 0 * 4096 + tid * 8;   // 16B per thread
        __bf16*       ldst = &sB[0][tid * 8];
        __builtin_amdgcn_global_load_async_to_lds_b128(
            (as1_v4i_ptr)(const void*)gsrc,
            (as3_v4i_ptr)(void*)ldst, 0, 0);
    }
#endif

    // ---- proj_h: sPH[u] = b1[u] + b2[u] + sum_h hidden[h]*W2[h,u] ----
    {
        float acc = b1[tid] + b2[tid];
#pragma unroll 4
        for (int h = 0; h < H_; ++h) acc += sH[h] * W2[h * U_ + tid];
        sPH[tid] = acc;
    }

    // ---- load A fragments (contiguous k order, same as the W1p pack):
    //   lane m (0-15):  VGPR r -> K = 2r      (+0/+1)
    //   lane 16+m:      VGPR r -> K = 16 + 2r
    const int m0 = wave << 4;
    const float* arow = fb + (m0 + m15) * D_;
    v16bf afrag[8];
#pragma unroll
    for (int kt = 0; kt < 8; ++kt) {
#pragma unroll
        for (int r = 0; r < 8; ++r) {
            int k = kt * 32 + (hi << 4) + (r << 1);
            float2 fv = *(const float2*)(arow + k);
            afrag[kt][2 * r]     = (__bf16)fv.x;
            afrag[kt][2 * r + 1] = (__bf16)fv.y;
        }
    }

#ifdef USE_ASYNC_LDS
    __builtin_amdgcn_s_wait_asynccnt(0);
#endif
    __syncthreads();   // sPH ready, slab 0 resident (async path)

    // per-lane logit partials: plog[v] belongs to L-row (m0 + v + 8*hi)
    float plog[8];
#pragma unroll
    for (int v = 0; v < 8; ++v) plog[v] = 0.0f;

    // ---- main GEMM sweep over 32 U-tiles ----
    for (int nt = 0; nt < 32; ++nt) {
        const int cur = nt & 1;
#ifdef USE_ASYNC_LDS
        if (nt + 1 < 32) {   // prefetch next slab into the other buffer
            const __bf16* gsrc = W1p + (nt + 1) * 4096 + tid * 8;
            __bf16*       ldst = &sB[cur ^ 1][tid * 8];
            __builtin_amdgcn_global_load_async_to_lds_b128(
                (as1_v4i_ptr)(const void*)gsrc,
                (as3_v4i_ptr)(void*)ldst, 0, 0);
        }
#else
        __syncthreads();  // previous slab fully consumed
        {   // coalesced 8KB slab copy: 512 threads x 16B
            const uint4* src = (const uint4*)(W1p + nt * 4096);
            ((uint4*)sB[0])[tid] = src[tid];
        }
        __syncthreads();
#endif
        const __bf16* slab =
#ifdef USE_ASYNC_LDS
            sB[cur];
#else
            sB[0];
#endif

        v8f acc = {0.f, 0.f, 0.f, 0.f, 0.f, 0.f, 0.f, 0.f};
#pragma unroll
        for (int kt = 0; kt < 8; ++kt) {
            v16bf bfrag = *(const v16bf*)(slab + kt * 512 + lane * 16);
            acc = __builtin_amdgcn_wmma_f32_16x16x32_bf16(
                false, afrag[kt], false, bfrag, (short)0, acc, false, false);
        }

        // ---- fused epilogue: accumulate tanh(acc + ph) * Wv in registers ----
        // C/D layout: VGPR v -> row (v + 8*hi), col N = lane&15.
        const int u  = (nt << 4) + m15;
        const float ph = sPH[u];
        const float wv = sWv[u];
#pragma unroll
        for (int v = 0; v < 8; ++v)
            plog[v] = fmaf(fast_tanh(acc[v] + ph), wv, plog[v]);

#ifdef USE_ASYNC_LDS
        __builtin_amdgcn_s_wait_asynccnt(0);  // next slab landed
        __syncthreads();                      // all waves done with sB[cur]
#endif
    }

    // ---- one-time butterfly reduction of logit partials over 16 columns ----
    // Each L-row is owned by exactly one (wave, half-wave): plain store.
#pragma unroll
    for (int v = 0; v < 8; ++v) {
        float p = plog[v];
        p += __shfl_xor(p, 1);
        p += __shfl_xor(p, 2);
        p += __shfl_xor(p, 4);
        p += __shfl_xor(p, 8);   // sum within each 16-lane half
        if (m15 == 0) sLogit[m0 + v + (hi << 3)] = p;
    }
    __syncthreads();

    // ---- softmax over L=256 logits (threads >= 256 contribute identity) ----
    float lg = (tid < L_) ? (sLogit[tid] + bv[0]) : -1e30f;
    float mx = lg;
#pragma unroll
    for (int off = 16; off > 0; off >>= 1) mx = fmaxf(mx, __shfl_xor(mx, off));
    if (lane == 0) sRed[wave] = mx;
    __syncthreads();
    if (tid == 0) {
        float m = sRed[0];
        for (int i = 1; i < 16; ++i) m = fmaxf(m, sRed[i]);
        sMax = m;
    }
    __syncthreads();
    float e = (tid < L_) ? __expf(lg - sMax) : 0.0f;
    float su = e;
#pragma unroll
    for (int off = 16; off > 0; off >>= 1) su += __shfl_xor(su, off);
    if (lane == 0) sRed[wave] = su;
    __syncthreads();
    if (tid == 0) {
        float s = 0.f;
        for (int i = 0; i < 16; ++i) s += sRed[i];
        sSum = s;
    }
    __syncthreads();
    if (tid < L_) {
        float w = e / sSum;
        sW[tid] = w;
        out[(size_t)B_ * D_ + (size_t)b * L_ + tid] = w;   // attention weights
    }
    __syncthreads();

    // ---- context vector: sum_l w[l] * features[b,l,:]  (coalesced, L2-hot) ----
    {
        const int d    = tid & (D_ - 1);
        const int part = tid >> 8;            // 0/1: split L range
        const float* fp = fb + part * (L_ / 2) * D_ + d;
        float acc = 0.0f;
#pragma unroll 4
        for (int l = 0; l < L_ / 2; ++l) acc += sW[part * (L_ / 2) + l] * fp[l * D_];
        sCtx[part][d] = acc;
    }
    __syncthreads();
    if (tid < D_) out[(size_t)b * D_ + tid] = sCtx[0][tid] + sCtx[1][tid];
}

// ---------------------------------------------------------------------------
extern "C" void kernel_launch(void* const* d_in, const int* in_sizes, int n_in,
                              void* d_out, int out_size, void* d_ws, size_t ws_size,
                              hipStream_t stream) {
    const float* features = (const float*)d_in[0];
    const float* hidden   = (const float*)d_in[1];
    const float* W1       = (const float*)d_in[2];
    const float* b1       = (const float*)d_in[3];
    const float* W2       = (const float*)d_in[4];
    const float* b2       = (const float*)d_in[5];
    const float* Wv       = (const float*)d_in[6];
    const float* bv       = (const float*)d_in[7];
    float* out = (float*)d_out;

    __bf16* W1p = (__bf16*)d_ws;   // 256 KB scratch: packed bf16 W1

    // Pack W1 into WMMA B-fragment layout: D*U = 131072 elements.
    hipLaunchKernelGGL(pack_w1_kernel, dim3((D_ * U_) / 256), dim3(256), 0, stream,
                       W1, W1p);
    // Fused attention: one WG per batch element.
    hipLaunchKernelGGL(attn_fused_kernel, dim3(B_), dim3(512), 0, stream,
                       features, hidden, b1, W2, b2, Wv, bv, W1p, out);
}
